// Net_5918464934506
// MI455X (gfx1250) — matrix operand order repaired and checked by reading, not compiled, not run on hardware
//
#include <hip/hip_runtime.h>
#include <hip/hip_bf16.h>

#define N_NODES 50000
#define E_EDGES 200000
#define F_IN    16
#define F_OUT   64
#define KN      32
#define NCHUNK  17          // 16 main K-chunks (K=512) + 1 bias chunk (b2 rows + zero pad)
#define TILE_E  16

typedef __attribute__((ext_vector_type(16))) _Float16 v16h;
typedef __attribute__((ext_vector_type(8)))  _Float16 v8h;
typedef __attribute__((ext_vector_type(8)))  float    v8f;

__device__ __forceinline__ void atomAddF32(float* p, float v) {
    unsafeAtomicAdd(p, v);   // native global_atomic_add_f32 on AMDGPU
}

// monotone float <-> uint mapping for atomicMax-based segment max
__device__ __forceinline__ unsigned f2ord(float f) {
    unsigned b = __float_as_uint(f);
    return (b & 0x80000000u) ? ~b : (b | 0x80000000u);
}
__device__ __forceinline__ float ord2f(unsigned u) {
    return (u & 0x80000000u) ? __uint_as_float(u & 0x7fffffffu) : __uint_as_float(~u);
}

// ---------------------------------------------------------------------------
// Kernel 0: zero the accumulation workspace (atomics require clean state)
// ---------------------------------------------------------------------------
__global__ void init_ws_kernel(float* __restrict__ agg, float* __restrict__ denom,
                               float* __restrict__ numer, unsigned* __restrict__ smax) {
    int i = blockIdx.x * blockDim.x + threadIdx.x;
    if (i < N_NODES * F_OUT) agg[i] = 0.f;
    if (i < N_NODES) { denom[i] = 0.f; numer[i] = 0.f; smax[i] = 0u; }
}

// ---------------------------------------------------------------------------
// Kernel 1: fused ECC dynamic-filter message + scatter-add  (WMMA f16)
//   A[e, k*16+f] = relu(e@W1+b1)[e,k] * x[src[e], f]   (packed-f16 outer product)
//   B[k*16+f, o] = W2[k, f*64+o]   (+16 bias rows from b2, zero pad to K=544)
//   msg = A @ B  -> atomicAdd into agg[dst]
// ---------------------------------------------------------------------------
__global__ __launch_bounds__(256) void ecc_msg_kernel(
    const float* __restrict__ x,  const float* __restrict__ e,
    const int*   __restrict__ esrc, const int* __restrict__ edst,
    const float* __restrict__ W1, const float* __restrict__ b1,
    const float* __restrict__ W2, const float* __restrict__ b2,
    float* __restrict__ agg, int nTiles)
{
    extern __shared__ char smemraw[];
    _Float16* Bsh  = (_Float16*)smemraw;                 // [NCHUNK][4][32 lanes][16 halfs]
    float*    W1sh = (float*)(smemraw + (size_t)NCHUNK * 4 * 32 * 16 * 2);
    float*    b1sh = W1sh + 4 * KN;

    const int tid = threadIdx.x;
    for (int i = tid; i < 4 * KN; i += blockDim.x) W1sh[i] = W1[i];
    if (tid < KN) b1sh[tid] = b1[tid];

    // Pre-swizzle B into per-lane WMMA fragment layout:
    // chunk c, n-tile t, lane = n + ((jj&8) ? 16:0), element = (jj&7) + ((jj&16) ? 8:0)  holds B[c*32+jj, t*16+n]
    const int totB = NCHUNK * 32 * F_OUT;
    for (int idx = tid; idx < totB; idx += blockDim.x) {
        int j = idx >> 6;          // K index 0..543
        int o = idx & 63;
        float v;
        if (j < 512)        { int k = j >> 4, f = j & 15; v = W2[k * (F_IN * F_OUT) + f * F_OUT + o]; }
        else if (j < 528)   { int f = j - 512;            v = b2[f * F_OUT + o]; }
        else                v = 0.f;
        int c = j >> 5, jj = j & 31;
        int t = o >> 4, n  = o & 15;
        int lane = n + ((jj & 8) ? 16 : 0);
        Bsh[((size_t)((c * 4 + t) * 32 + lane)) * 16 + ((jj & 7) + ((jj & 16) ? 8 : 0))] = (_Float16)v;
    }
    __syncthreads();

    const int lane = tid & 31;
    const int wave = tid >> 5;
    const int tile = blockIdx.x * (blockDim.x >> 5) + wave;
    if (tile >= nTiles) return;                 // uniform per wave: EXEC stays all-1 for WMMA
    const int ebase = tile * TILE_E;
    const int m  = lane & 15;                   // A-matrix row owned by this lane
    const int hi = lane >> 4;                   // half-lane selector
    const int hb = hi * 8;                      // f-offset of this lane's 8 elements

    // --- kernel-network layer 1: h = relu(e @ W1 + b1), one edge row per lane ---
    // Accumulate in f32, then keep the 32-entry row live as packed f16 (16 VGPRs).
    float ef[4];
    #pragma unroll
    for (int s = 0; s < 4; ++s) ef[s] = e[(ebase + m) * 4 + s];
    _Float16 hh[KN];
    #pragma unroll
    for (int k = 0; k < KN; ++k) {
        float a = b1sh[k];
        #pragma unroll
        for (int s = 0; s < 4; ++s) a += ef[s] * W1sh[s * KN + k];
        hh[k] = (_Float16)(a > 0.f ? a : 0.f);
    }

    // --- gather source-node features this lane contributes (convert to f16 once) ---
    const int src = esrc[ebase + m];
    v8h xh;
    #pragma unroll
    for (int i = 0; i < 8; ++i) xh[i] = (_Float16)x[src * F_IN + hb + i];

    v8f acc[4] = { v8f{}, v8f{}, v8f{}, v8f{} };

    // --- 16 main K-chunks: A = [ hh[2c]*xh | hh[2c+1]*xh ] via v_pk_mul_f16 ---
    #pragma unroll
    for (int c = 0; c < 16; ++c) {
        v8h lo = xh * hh[2 * c];
        v8h hi8 = xh * hh[2 * c + 1];
        v16h A = __builtin_shufflevector(lo, hi8,
                                         0, 1, 2, 3, 4, 5, 6, 7,
                                         8, 9, 10, 11, 12, 13, 14, 15);
        #pragma unroll
        for (int t = 0; t < 4; ++t) {
            v16h B = *(const v16h*)&Bsh[((size_t)((c * 4 + t) * 32 + lane)) * 16];
            acc[t] = __builtin_amdgcn_wmma_f32_16x16x32_f16(
                false, A, false, B, (short)0, acc[t], false, false);
        }
    }
    // --- bias chunk (c=16): rows 512..527 carry x[f] (to apply b2), 528..543 zero ---
    {
        v8h zero = (v8h)(_Float16)0.f;
        v16h A = __builtin_shufflevector(xh, zero,
                                         0, 1, 2, 3, 4, 5, 6, 7,
                                         8, 9, 10, 11, 12, 13, 14, 15);
        #pragma unroll
        for (int t = 0; t < 4; ++t) {
            v16h B = *(const v16h*)&Bsh[((size_t)((16 * 4 + t) * 32 + lane)) * 16];
            acc[t] = __builtin_amdgcn_wmma_f32_16x16x32_f16(
                false, A, false, B, (short)0, acc[t], false, false);
        }
    }

    // --- scatter-add: D element r -> edge (r + hi*8), column t*16 + (lane&15) ---
    const int n = lane & 15;
    #pragma unroll
    for (int r = 0; r < 8; ++r) {
        const int d = edst[ebase + r + hi * 8];
        #pragma unroll
        for (int t = 0; t < 4; ++t)
            atomAddF32(&agg[(size_t)d * F_OUT + t * 16 + n], acc[t][r]);
    }
}

// ---------------------------------------------------------------------------
// Kernel 2: x1 = relu(agg + x@root_kernel + ecc_bias); xp = x1 @ gat_kernel
// ---------------------------------------------------------------------------
__global__ __launch_bounds__(256) void node_update_kernel(
    const float* __restrict__ x, const float* __restrict__ agg,
    const float* __restrict__ root_kernel, const float* __restrict__ ecc_bias,
    const float* __restrict__ gat_kernel, float* __restrict__ xp)
{
    __shared__ float rk[F_IN * F_OUT];
    __shared__ float eb[F_OUT];
    __shared__ float gk[F_OUT];
    for (int i = threadIdx.x; i < F_IN * F_OUT; i += blockDim.x) rk[i] = root_kernel[i];
    if (threadIdx.x < F_OUT) { eb[threadIdx.x] = ecc_bias[threadIdx.x]; gk[threadIdx.x] = gat_kernel[threadIdx.x]; }
    __syncthreads();

    const int nid = blockIdx.x * blockDim.x + threadIdx.x;
    if (nid >= N_NODES) return;
    float xf[F_IN];
    #pragma unroll
    for (int f = 0; f < F_IN; ++f) xf[f] = x[(size_t)nid * F_IN + f];
    float s = 0.f;
    for (int o = 0; o < F_OUT; ++o) {
        float a = agg[(size_t)nid * F_OUT + o] + eb[o];
        #pragma unroll
        for (int f = 0; f < F_IN; ++f) a += xf[f] * rk[f * F_OUT + o];
        a = a > 0.f ? a : 0.f;
        s += a * gk[o];
    }
    xp[nid] = s;
}

// ---------------------------------------------------------------------------
// Kernel 3: segment max of leaky_relu scores over E edges + N self loops
// ---------------------------------------------------------------------------
__global__ void score_max_kernel(const int* __restrict__ esrc, const int* __restrict__ edst,
                                 const float* __restrict__ xp,
                                 const float* __restrict__ attn_self,
                                 const float* __restrict__ attn_neigh,
                                 unsigned* __restrict__ smax)
{
    const int i = blockIdx.x * blockDim.x + threadIdx.x;
    if (i >= E_EDGES + N_NODES) return;
    const float as = attn_self[0], an = attn_neigh[0];
    int s, d;
    if (i < E_EDGES) { s = esrc[i]; d = edst[i]; } else { s = d = i - E_EDGES; }
    float sc = xp[d] * as + xp[s] * an;
    sc = sc > 0.f ? sc : 0.2f * sc;
    atomicMax(&smax[d], f2ord(sc));
}

// ---------------------------------------------------------------------------
// Kernel 4: ex = exp(score - smax[dst]);  denom[d] += ex;  numer[d] += ex*xp[src]
// ---------------------------------------------------------------------------
__global__ void attn_acc_kernel(const int* __restrict__ esrc, const int* __restrict__ edst,
                                const float* __restrict__ xp,
                                const float* __restrict__ attn_self,
                                const float* __restrict__ attn_neigh,
                                const unsigned* __restrict__ smax,
                                float* __restrict__ denom, float* __restrict__ numer)
{
    const int i = blockIdx.x * blockDim.x + threadIdx.x;
    if (i >= E_EDGES + N_NODES) return;
    const float as = attn_self[0], an = attn_neigh[0];
    int s, d;
    if (i < E_EDGES) { s = esrc[i]; d = edst[i]; } else { s = d = i - E_EDGES; }
    float sc = xp[d] * as + xp[s] * an;
    sc = sc > 0.f ? sc : 0.2f * sc;
    const float ex = __expf(sc - ord2f(smax[d]));
    atomAddF32(&denom[d], ex);
    atomAddF32(&numer[d], ex * xp[s]);
}

// ---------------------------------------------------------------------------
// Kernel 5: out = sigmoid(numer/denom + gat_bias)
// ---------------------------------------------------------------------------
__global__ void out_kernel(const float* __restrict__ numer, const float* __restrict__ denom,
                           const float* __restrict__ gat_bias, float* __restrict__ out)
{
    const int i = blockIdx.x * blockDim.x + threadIdx.x;
    if (i >= N_NODES) return;
    const float v = numer[i] / denom[i] + gat_bias[0];
    out[i] = 1.f / (1.f + __expf(-v));
}

// ---------------------------------------------------------------------------
extern "C" void kernel_launch(void* const* d_in, const int* in_sizes, int n_in,
                              void* d_out, int out_size, void* d_ws, size_t ws_size,
                              hipStream_t stream) {
    const float* x     = (const float*)d_in[0];
    const float* e     = (const float*)d_in[1];
    const int*   esrc  = (const int*)  d_in[2];
    const int*   edst  = (const int*)  d_in[3];
    const float* W1    = (const float*)d_in[4];
    const float* b1    = (const float*)d_in[5];
    const float* W2    = (const float*)d_in[6];
    const float* b2    = (const float*)d_in[7];
    const float* rootk = (const float*)d_in[8];
    const float* eccb  = (const float*)d_in[9];
    const float* gatk  = (const float*)d_in[10];
    const float* atts  = (const float*)d_in[11];
    const float* attn  = (const float*)d_in[12];
    const float* gatb  = (const float*)d_in[13];
    float* out = (float*)d_out;

    char* ws = (char*)d_ws;
    size_t off = 0;
    auto carve = [&](size_t bytes) { void* p = ws + off; off = (off + bytes + 255) & ~(size_t)255; return p; };
    float*    agg   = (float*)   carve((size_t)N_NODES * F_OUT * sizeof(float));
    float*    xp    = (float*)   carve((size_t)N_NODES * sizeof(float));
    unsigned* smax  = (unsigned*)carve((size_t)N_NODES * sizeof(unsigned));
    float*    denom = (float*)   carve((size_t)N_NODES * sizeof(float));
    float*    numer = (float*)   carve((size_t)N_NODES * sizeof(float));

    // 0) zero accumulators
    init_ws_kernel<<<(N_NODES * F_OUT + 255) / 256, 256, 0, stream>>>(agg, denom, numer, smax);

    // 1) fused ECC message GEMM (WMMA) + scatter
    const int nTiles = E_EDGES / TILE_E;                // 12500, exact
    const int wavesPerBlock = 256 / 32;                 // 8
    const int blocks = (nTiles + wavesPerBlock - 1) / wavesPerBlock;
    const size_t shbytes = (size_t)NCHUNK * 4 * 32 * 16 * 2   // B fragments (f16)
                         + (4 * KN + KN) * sizeof(float);     // W1 + b1
    ecc_msg_kernel<<<blocks, 256, shbytes, stream>>>(x, e, esrc, edst, W1, b1, W2, b2, agg, nTiles);

    // 2) node update -> xp
    node_update_kernel<<<(N_NODES + 255) / 256, 256, 0, stream>>>(x, agg, rootk, eccb, gatk, xp);

    // 3) segment max
    score_max_kernel<<<(E_EDGES + N_NODES + 255) / 256, 256, 0, stream>>>(esrc, edst, xp, atts, attn, smax);

    // 4) exp + accumulate numer/denom
    attn_acc_kernel<<<(E_EDGES + N_NODES + 255) / 256, 256, 0, stream>>>(esrc, edst, xp, atts, attn, smax, denom, numer);

    // 5) normalize + sigmoid
    out_kernel<<<(N_NODES + 255) / 256, 256, 0, stream>>>(numer, denom, gatb, out);
}
